// SelfConditioningResidualLayer_63393717289730
// MI455X (gfx1250) — matrix operand order repaired and checked by reading, not compiled, not run on hardware
//
#include <hip/hip_runtime.h>
#include <hip/hip_bf16.h>

typedef __attribute__((ext_vector_type(16))) _Float16 v16h;
typedef __attribute__((ext_vector_type(8)))  float    v8f;

union FragU  { uint4 q[2]; v16h h; };
union StageU { v16h h; uint4 q[2]; };

// ---------------------------------------------------------------------------
// constants
// ---------------------------------------------------------------------------
#define NNODE   100000
#define EHALF   250000
#define NPAD    100096      // round up to 128
#define EPAD    250112      // round up to 128
#define D_NODE  256
#define D_EDGE  128
#define KN1     320         // 296 padded to 32
#define KE1     160         // 148 padded to 32

#define BM      128
#define BN      128
#define BK      32
#define LDA_S   40          // LDS A row stride (f16), padded for bank spread

#define MODE_NODE 0
#define MODE_EDGE 1

// ---------------------------------------------------------------------------
// weight conversion: f32 [K x Ncols] -> f16 [Kpad x Ncols], zero pad rows
// ---------------------------------------------------------------------------
__global__ void cvt_weights(const float* __restrict__ W, _Float16* __restrict__ Wh,
                            int K, int Kpad, int Ncols) {
    int idx = blockIdx.x * 256 + threadIdx.x;
    if (idx >= Kpad * Ncols) return;
    int r = idx / Ncols, c = idx - r * Ncols;
    Wh[idx] = (r < K) ? (_Float16)W[r * Ncols + c] : (_Float16)0.0f;
}

__device__ __forceinline__ float pair_norm(const float* p, int a, int b) {
    float dx = p[a * 3 + 0] - p[b * 3 + 0];
    float dy = p[a * 3 + 1] - p[b * 3 + 1];
    float dz = p[a * 3 + 2] - p[b * 3 + 2];
    return sqrtf(dx * dx + dy * dy + dz * dz + 1e-8f);
}

// ---------------------------------------------------------------------------
// on-the-fly A-tile staging: 16 f16 input-features [c0, c0+16) of global row
// ---------------------------------------------------------------------------
template <int MODE>
__device__ __forceinline__ void stage_vals(
    StageU& sv, size_t grow, bool valid, int c0,
    const float* __restrict__ srcMain,            // s_t / e_t
    const float* __restrict__ aux0,               // a_dst / e_dst
    const float* __restrict__ aux1,               // c_dst / unused
    const float* __restrict__ x_t, const float* __restrict__ x_dst,
    const int* __restrict__ esrc, const int* __restrict__ edst)
{
    constexpr int MAINW = (MODE == MODE_NODE) ? 256 : 128;
    if (!valid) {
#pragma unroll
        for (int e = 0; e < 16; ++e) sv.h[e] = (_Float16)0.0f;
        return;
    }
    if (c0 + 16 <= MAINW) {                       // pure main-feature chunk
        const float4* p = (const float4*)(srcMain + grow * MAINW + c0);
#pragma unroll
        for (int q = 0; q < 4; ++q) {
            float4 f = p[q];
            sv.h[q * 4 + 0] = (_Float16)f.x; sv.h[q * 4 + 1] = (_Float16)f.y;
            sv.h[q * 4 + 2] = (_Float16)f.z; sv.h[q * 4 + 3] = (_Float16)f.w;
        }
        return;
    }
    // mixed tail (concat features + RBF + zero pad)
    float d0 = 0.0f, d1 = 0.0f;
    if constexpr (MODE == MODE_NODE) {
        float dx = x_t[grow * 3 + 0] - x_dst[grow * 3 + 0];
        float dy = x_t[grow * 3 + 1] - x_dst[grow * 3 + 1];
        float dz = x_t[grow * 3 + 2] - x_dst[grow * 3 + 2];
        d0 = sqrtf(dx * dx + dy * dy + dz * dz + 1e-8f);
    } else {
        int s = esrc[grow], t = edst[grow];
        d0 = pair_norm(x_t,  s, t) + 1e-8f;
        d1 = pair_norm(x_dst, s, t) + 1e-8f;
    }
#pragma unroll
    for (int e = 0; e < 16; ++e) {
        const int col = c0 + e;
        float v = 0.0f;
        if constexpr (MODE == MODE_NODE) {
            if (col < 256)      v = srcMain[grow * 256 + col];
            else if (col < 272) v = aux0[grow * 16 + (col - 256)];
            else if (col < 280) v = aux1[grow * 8 + (col - 272)];
            else if (col < 296) {
                float mu = 0.66666667f * (float)(col - 280);   // linspace(0,10,16)
                float z  = (d0 - mu) * 1.6f;                   // sigma = 10/16
                v = __expf(-z * z);
            }
        } else {
            if (col < 128)      v = srcMain[grow * 128 + col];
            else if (col < 132) v = aux0[grow * 4 + (col - 128)];
            else if (col < 148) {
                float mu = 0.66666667f * (float)(col - 132);
                float z1 = (d1 - mu) * 1.6f;
                float z0 = (d0 - mu) * 1.6f;
                v = __expf(-z1 * z1) - __expf(-z0 * z0);
            }
        }
        sv.h[e] = (_Float16)v;
    }
}

// ---------------------------------------------------------------------------
// fully fused 2-layer MLP:
//   out = residual + silu(silu(build(row) @ W1 + b1) @ W2 + b2)
// Block = 128 rows. Hidden tile lives in LDS; A tile built on the fly from
// raw inputs; B fragments read straight from L2-resident f16 weights.
// 8 waves: 4 in M x 2 in N, each wave 32x64 via 2x4 WMMA f32_16x16x32_f16.
// ---------------------------------------------------------------------------
template <int MODE, int KIN, int D>
__global__ __launch_bounds__(256) void fused_mlp(
    const float* __restrict__ srcMain,            // s_t / e_t (also residual)
    const float* __restrict__ aux0,               // a_dst / e_dst
    const float* __restrict__ aux1,               // c_dst / unused
    const float* __restrict__ x_t, const float* __restrict__ x_dst,
    const int* __restrict__ esrc, const int* __restrict__ edst,
    const _Float16* __restrict__ W1, const float* __restrict__ b1,
    const _Float16* __restrict__ W2, const float* __restrict__ b2,
    float* __restrict__ Cout, float* __restrict__ Cout2,
    int Mreal)
{
    extern __shared__ char smem[];
    _Float16* As = (_Float16*)smem;                            // 2*BM*LDA_S
    _Float16* Hs = (_Float16*)(smem + 2 * BM * LDA_S * 2);     // BM*(D+8)
    constexpr int HLD = D + 8;

    const int tid  = threadIdx.x;
    const int lane = tid & 31;
    const int wave = tid >> 5;
    const int wm   = wave & 3;
    const int wn   = wave >> 2;
    const size_t m0 = (size_t)blockIdx.x * BM;

    const int ar = tid >> 1;               // staged row 0..127
    const int ah = (tid & 1) << 4;         // chunk 0 / 16
    const size_t grow = m0 + ar;
    const bool valid = grow < (size_t)Mreal;
    const int k0 = (lane < 16) ? 0 : 8;    // A-fragment K sub-chunk select

    if (tid == 0) { __builtin_prefetch(W1, 0, 1); __builtin_prefetch(W2, 0, 1); }

    // ---------------- layer 1: build A on the fly, H -> LDS ----------------
#pragma unroll
    for (int nb = 0; nb < D / BN; ++nb) {
        v8f acc[2][4] = {};
        constexpr int NK = KIN / 32;
        StageU sv;
        stage_vals<MODE>(sv, grow, valid, ah, srcMain, aux0, aux1, x_t, x_dst, esrc, edst);
#pragma unroll
        for (int ks = 0; ks < NK; ++ks) {
            uint4* o = (uint4*)&As[(ks & 1) * BM * LDA_S + ar * LDA_S + ah];
            o[0] = sv.q[0]; o[1] = sv.q[1];
            __syncthreads();
            if (ks + 1 < NK)
                stage_vals<MODE>(sv, grow, valid, (ks + 1) * 32 + ah,
                                 srcMain, aux0, aux1, x_t, x_dst, esrc, edst);
            const _Float16* Ab = &As[(ks & 1) * BM * LDA_S];
            FragU fa[2], fb[4];
#pragma unroll
            for (int mi = 0; mi < 2; ++mi) {
                const _Float16* p = Ab + (wm * 32 + mi * 16 + (lane & 15)) * LDA_S + k0;
                fa[mi].q[0] = *(const uint4*)(p);
                fa[mi].q[1] = *(const uint4*)(p + 16);
            }
#pragma unroll
            for (int ni = 0; ni < 4; ++ni) {   // B frag: lane = N col, 16 K values
                const _Float16* p = W1 + (size_t)(ks * 32 + k0) * D
                                       + nb * BN + wn * 64 + ni * 16 + (lane & 15);
#pragma unroll
                for (int s = 0; s < 16; ++s)
                    fb[ni].h[s] = p[(size_t)((s & 7) + ((s >> 3) << 4)) * D];
            }
#pragma unroll
            for (int mi = 0; mi < 2; ++mi)
#pragma unroll
                for (int ni = 0; ni < 4; ++ni)
                    acc[mi][ni] = __builtin_amdgcn_wmma_f32_16x16x32_f16(
                        false, fa[mi].h, false, fb[ni].h,
                        (short)0, acc[mi][ni], false, false);
        }
        // epilogue: bias + SiLU -> LDS hidden tile (f16)
#pragma unroll
        for (int mi = 0; mi < 2; ++mi)
#pragma unroll
            for (int ni = 0; ni < 4; ++ni) {
                const int colL = nb * BN + wn * 64 + ni * 16 + (lane & 15);
                const float bv = b1[colL];
                const int rl0  = wm * 32 + mi * 16 + ((lane >> 4) << 3);
#pragma unroll
                for (int r = 0; r < 8; ++r) {
                    float v = acc[mi][ni][r] + bv;
                    float s = v / (1.0f + __expf(-v));
                    Hs[(rl0 + r) * HLD + colL] = (_Float16)s;
                }
            }
        __syncthreads();
    }

    // ---------------- layer 2: A straight from LDS hidden tile -------------
#pragma unroll
    for (int nb = 0; nb < D / BN; ++nb) {
        v8f acc[2][4] = {};
        constexpr int NK = D / 32;
#pragma unroll
        for (int ks = 0; ks < NK; ++ks) {
            FragU fa[2], fb[4];
#pragma unroll
            for (int mi = 0; mi < 2; ++mi) {
                const _Float16* p = Hs + (wm * 32 + mi * 16 + (lane & 15)) * HLD
                                       + ks * 32 + k0;
                fa[mi].q[0] = *(const uint4*)(p);
                fa[mi].q[1] = *(const uint4*)(p + 16);
            }
#pragma unroll
            for (int ni = 0; ni < 4; ++ni) {
                const _Float16* p = W2 + (size_t)(ks * 32 + k0) * D
                                       + nb * BN + wn * 64 + ni * 16 + (lane & 15);
#pragma unroll
                for (int s = 0; s < 16; ++s)
                    fb[ni].h[s] = p[(size_t)((s & 7) + ((s >> 3) << 4)) * D];
            }
#pragma unroll
            for (int mi = 0; mi < 2; ++mi)
#pragma unroll
                for (int ni = 0; ni < 4; ++ni)
                    acc[mi][ni] = __builtin_amdgcn_wmma_f32_16x16x32_f16(
                        false, fa[mi].h, false, fb[ni].h,
                        (short)0, acc[mi][ni], false, false);
        }
        // epilogue: bias + SiLU + residual (+ duplicate for edge) -> f32 out
#pragma unroll
        for (int mi = 0; mi < 2; ++mi)
#pragma unroll
            for (int ni = 0; ni < 4; ++ni) {
                const int colL = nb * BN + wn * 64 + ni * 16 + (lane & 15);
                const float bv = b2[colL];
                const size_t rb = m0 + wm * 32 + mi * 16 + (size_t)((lane >> 4) << 3);
#pragma unroll
                for (int r = 0; r < 8; ++r) {
                    const size_t row = rb + r;
                    if (row < (size_t)Mreal) {
                        float v = acc[mi][ni][r] + bv;
                        float s = v / (1.0f + __expf(-v));
                        const size_t idx = row * (size_t)D + colL;
                        float o = srcMain[idx] + s;
                        Cout[idx] = o;
                        if constexpr (MODE == MODE_EDGE) Cout2[idx] = o;
                    }
                }
            }
    }
}

// ---------------------------------------------------------------------------
// launcher
// ---------------------------------------------------------------------------
extern "C" void kernel_launch(void* const* d_in, const int* in_sizes, int n_in,
                              void* d_out, int out_size, void* d_ws, size_t ws_size,
                              hipStream_t stream) {
    (void)in_sizes; (void)n_in; (void)out_size; (void)ws_size;

    const float* s_t   = (const float*)d_in[0];
    const float* x_t   = (const float*)d_in[1];
    const float* v_t   = (const float*)d_in[2];
    const float* e_t   = (const float*)d_in[3];
    const float* x_dst = (const float*)d_in[4];
    const float* a_dst = (const float*)d_in[5];
    const float* c_dst = (const float*)d_in[6];
    const float* e_dst = (const float*)d_in[7];
    const float* Wn1   = (const float*)d_in[8];
    const float* bn1   = (const float*)d_in[9];
    const float* Wn2   = (const float*)d_in[10];
    const float* bn2   = (const float*)d_in[11];
    const float* We1   = (const float*)d_in[12];
    const float* be1   = (const float*)d_in[13];
    const float* We2   = (const float*)d_in[14];
    const float* be2   = (const float*)d_in[15];
    const int*   esrc  = (const int*)d_in[16];
    const int*   edst  = (const int*)d_in[17];

    // workspace: only the converted f16 weights (no HBM activations anymore)
    char* ws = (char*)d_ws;
    _Float16* Wn1h = (_Float16*)(ws + 0);          // 320*256*2 = 163840
    _Float16* Wn2h = (_Float16*)(ws + 163840);     // 256*256*2 = 131072
    _Float16* We1h = (_Float16*)(ws + 294912);     // 160*128*2 =  40960
    _Float16* We2h = (_Float16*)(ws + 335872);     // 128*128*2 =  32768

    float* out_node = (float*)d_out;                        // N x 256
    float* out_x    = out_node + (size_t)NNODE * 256;       // N x 3
    float* out_v    = out_x    + (size_t)NNODE * 3;         // N x 3
    float* out_edge = out_v    + (size_t)NNODE * 3;         // 2*EHALF x 128

    cvt_weights<<<(KN1 * D_NODE + 255) / 256, 256, 0, stream>>>(Wn1, Wn1h, 296, KN1, D_NODE);
    cvt_weights<<<(D_NODE * D_NODE + 255) / 256, 256, 0, stream>>>(Wn2, Wn2h, 256, 256, D_NODE);
    cvt_weights<<<(KE1 * D_EDGE + 255) / 256, 256, 0, stream>>>(We1, We1h, 148, KE1, D_EDGE);
    cvt_weights<<<(D_EDGE * D_EDGE + 255) / 256, 256, 0, stream>>>(We2, We2h, 128, 128, D_EDGE);

    // node path: one fused kernel, LDS = A dbl-buf (20.5KB) + H tile (66KB)
    {
        size_t smem = 2 * BM * LDA_S * 2 + (size_t)BM * (D_NODE + 8) * 2;  // 88064
        fused_mlp<MODE_NODE, KN1, D_NODE><<<NPAD / BM, 256, smem, stream>>>(
            s_t, a_dst, c_dst, x_t, x_dst, nullptr, nullptr,
            Wn1h, bn1, Wn2h, bn2, out_node, nullptr, NNODE);
    }

    // position / velocity pass-through
    hipMemcpyAsync(out_x, x_t, (size_t)NNODE * 3 * sizeof(float), hipMemcpyDeviceToDevice, stream);
    hipMemcpyAsync(out_v, v_t, (size_t)NNODE * 3 * sizeof(float), hipMemcpyDeviceToDevice, stream);

    // edge path: one fused kernel, LDS = 20.5KB + 34KB
    {
        size_t smem = 2 * BM * LDA_S * 2 + (size_t)BM * (D_EDGE + 8) * 2;  // 55296
        fused_mlp<MODE_EDGE, KE1, D_EDGE><<<EPAD / BM, 256, smem, stream>>>(
            e_t, e_dst, nullptr, x_t, x_dst, esrc, edst,
            We1h, be1, We2h, be2, out_edge, out_edge + (size_t)EHALF * 128, EHALF);
    }
}
